// FusionKANLayer_4827543240871
// MI455X (gfx1250) — compile-verified, matrix operand order checked by reading, not run on hardware
//
#include <hip/hip_runtime.h>
#include <stdint.h>

#define B_DIM 4096
#define IN_DIM 2048
#define OUT_DIM 2048
#define NCOEF 9                    // 1 silu + 8 spline bases per input feature
#define K_DIM (IN_DIM * NCOEF)     // 18432
#define KB 64                      // K elements per pipeline stage
#define AST 72                     // padded LDS row stride in bf16 elems (144 B, 16B-aligned)

typedef __attribute__((ext_vector_type(16))) __bf16 v16bf;
typedef __attribute__((ext_vector_type(8)))  float  v8f;

__device__ __forceinline__ uint16_t f2bf(float f) {
  uint32_t u = __float_as_uint(f);
  uint32_t r = (u + 0x7FFFu + ((u >> 16) & 1u)) >> 16;   // round-to-nearest-even
  return (uint16_t)r;
}

// -------- CDNA5 async global->LDS copy (16B per lane), tracked by ASYNCcnt
__device__ __forceinline__ void async_copy_b128(const uint16_t* gp, uint16_t* lp) {
  uint32_t l = (uint32_t)(uintptr_t)lp;          // LDS byte offset (addr[31:0])
  uint64_t g = (uint64_t)(uintptr_t)gp;
  asm volatile("global_load_async_to_lds_b128 %0, %1, off"
               :: "v"(l), "v"(g) : "memory");
}
__device__ __forceinline__ void wait_async0() {
  asm volatile("s_wait_asynccnt 0x0" ::: "memory");
}

// ---------------------------------------------------------------- batch stats
__global__ __launch_bounds__(256) void kan_colstats(const float* __restrict__ x,
                                                    float* __restrict__ mu,
                                                    float* __restrict__ rstd) {
  __shared__ float ssum[8][32];
  __shared__ float ssq[8][32];
  int tx = threadIdx.x;                 // column lane 0..31 (coalesced)
  int ty = threadIdx.y;                 // row group 0..7
  int col = blockIdx.x * 32 + tx;
  float s = 0.f, q = 0.f;
  for (int r = ty; r < B_DIM; r += 8) {
    float v = x[(size_t)r * IN_DIM + col];
    s += v; q += v * v;
  }
  ssum[ty][tx] = s; ssq[ty][tx] = q;
  __syncthreads();
  if (ty == 0) {
    for (int j = 1; j < 8; ++j) { s += ssum[j][tx]; q += ssq[j][tx]; }
    float m = s * (1.f / B_DIM);
    float var = q * (1.f / B_DIM) - m * m;
    mu[col] = m;
    rstd[col] = rsqrtf(var + 1e-5f);
  }
}

// ------------------------------------------- BN + SiLU + B-spline bases (bf16)
__global__ __launch_bounds__(256) void kan_build_act(const float* __restrict__ x,
    const float* __restrict__ mu, const float* __restrict__ rstd,
    const float* __restrict__ gamma, const float* __restrict__ beta,
    uint16_t* __restrict__ A) {
  int idx = blockIdx.x * blockDim.x + threadIdx.x;
  int b = idx / IN_DIM, i = idx % IN_DIM;
  float xv = x[idx];
  float xn = (xv - mu[i]) * rstd[i] * gamma[i] + beta[i];
  float s = xn / (1.f + __expf(-xn));                 // SiLU
  // Cox-de Boor, uniform knots t[j] = -3 + 1.2*(j-3), j = 0..11
  const float h = 1.2f;
  float t[12];
#pragma unroll
  for (int j = 0; j < 12; ++j) t[j] = -3.f + h * (float)(j - 3);
  float bb[11];
#pragma unroll
  for (int j = 0; j < 11; ++j) bb[j] = (xn >= t[j] && xn < t[j + 1]) ? 1.f : 0.f;
#pragma unroll
  for (int k = 1; k <= 3; ++k) {
    float inv = 1.f / (h * (float)k);                 // uniform: t[j+k]-t[j] = k*h
#pragma unroll
    for (int j = 0; j < 11 - k; ++j)
      bb[j] = (xn - t[j]) * inv * bb[j] + (t[j + k + 1] - xn) * inv * bb[j + 1];
  }
  uint16_t* p = A + (size_t)b * K_DIM + (size_t)i * NCOEF;
  p[0] = f2bf(s);
#pragma unroll
  for (int c = 0; c < 8; ++c) p[1 + c] = f2bf(bb[c]);
}

// ---------------------------------- fused bf16 weight matrix + folded-in scales
__global__ __launch_bounds__(256) void kan_build_w(const float* __restrict__ bw,
    const float* __restrict__ sw, const float* __restrict__ sb,
    const float* __restrict__ ss, const float* __restrict__ bias,
    uint16_t* __restrict__ W, float* __restrict__ ybias) {
  int idx = blockIdx.x * blockDim.x + threadIdx.x;
  int o = idx / IN_DIM, i = idx % IN_DIM;
  float sbv = sb[o], ssv = ss[o];
  uint16_t* p = W + (size_t)o * K_DIM + (size_t)i * NCOEF;
  p[0] = f2bf(bw[idx] * sbv);
  const float* sp = sw + (size_t)idx * 8;
#pragma unroll
  for (int c = 0; c < 8; ++c) p[1 + c] = f2bf(sp[c] * ssv);
  if (i == 0) ybias[o] = sbv * bias[o];
}

// -------------------- 16-bit WMMA fragment from LDS (ISA 16x32 A/B VGPR layout)
// lane L (0..15 -> M/N = L, K half 0; 16..31 -> same M/N, K half +8).
// VGPR pairs: group0 = K[hi8 .. hi8+7], group1 = K[16+hi8 .. 16+hi8+7].
__device__ __forceinline__ v16bf load_frag(const uint16_t* row0, int lane) {
  int r = lane & 15;
  int hi = (lane >> 4) << 3;
  const uint16_t* rp = row0 + r * AST;
  union { uint4 u[2]; v16bf v; } u;
  u.u[0] = *(const uint4*)(rp + hi);        // ds_load_b128
  u.u[1] = *(const uint4*)(rp + 16 + hi);   // ds_load_b128
  return u.v;
}

// ----------------------------- fused GEMM: Y = A(4096xK) * W(2048xK)^T + ybias
// 128x128 block tile, 8 waves (wave32), each wave 32x64 -> 8 WMMA accums.
// Double-buffered LDS, async global->LDS pipeline, one barrier per K-stage.
__global__ __launch_bounds__(256) void kan_gemm(const uint16_t* __restrict__ A,
    const uint16_t* __restrict__ W, const float* __restrict__ ybias,
    float* __restrict__ Y) {
  __shared__ __align__(16) uint16_t As[2][128 * AST];
  __shared__ __align__(16) uint16_t Bs[2][128 * AST];
  int tid  = threadIdx.x;
  int lane = tid & 31, wave = tid >> 5;
  int wm = wave & 3;                 // wave M subtile: rows wm*32..wm*32+31
  int wn = wave >> 2;                // wave N subtile: cols wn*64..wn*64+63
  int bm = blockIdx.x * 128;
  int bn = blockIdx.y * 128;
  v8f acc[2][4] = {};

  // async staging map: thread copies 16B chunks; rows r0+32j, cols c0..c0+7
  int r0 = tid >> 3;                 // 0..31
  int c0 = (tid & 7) * 8;            // 0,8,...,56
  const uint16_t* ag = A + (size_t)(bm + r0) * K_DIM + c0;
  const uint16_t* wg = W + (size_t)(bn + r0) * K_DIM + c0;
  const size_t rowq = (size_t)32 * K_DIM;
  int sa = r0 * AST + c0;

  // prologue: stage tile 0 into buffer 0
#pragma unroll
  for (int j = 0; j < 4; ++j) {
    async_copy_b128(ag + j * rowq, &As[0][sa + j * 32 * AST]);
    async_copy_b128(wg + j * rowq, &Bs[0][sa + j * 32 * AST]);
  }

  const int NSTAGE = K_DIM / KB;     // 288
  int buf = 0;
  for (int k = 0; k < NSTAGE; ++k) {
    wait_async0();                   // my stage-k copies have landed
    __syncthreads();                 // everyone's landed; stage k-1 reads done
    if (k + 1 < NSTAGE) {            // prefetch stage k+1 into the other buffer
      const uint16_t* agn = ag + (size_t)(k + 1) * KB;
      const uint16_t* wgn = wg + (size_t)(k + 1) * KB;
#pragma unroll
      for (int j = 0; j < 4; ++j) {
        async_copy_b128(agn + j * rowq, &As[buf ^ 1][sa + j * 32 * AST]);
        async_copy_b128(wgn + j * rowq, &Bs[buf ^ 1][sa + j * 32 * AST]);
      }
    }
#pragma unroll
    for (int ks = 0; ks < 2; ++ks) { // two 32-K WMMA chunks per stage
      v16bf af[2], bf_[4];
#pragma unroll
      for (int mi = 0; mi < 2; ++mi)
        af[mi] = load_frag(&As[buf][(wm * 32 + mi * 16) * AST + ks * 32], lane);
#pragma unroll
      for (int ni = 0; ni < 4; ++ni)
        bf_[ni] = load_frag(&Bs[buf][(wn * 64 + ni * 16) * AST + ks * 32], lane);
#pragma unroll
      for (int mi = 0; mi < 2; ++mi)
#pragma unroll
        for (int ni = 0; ni < 4; ++ni)
          acc[mi][ni] = __builtin_amdgcn_wmma_f32_16x16x32_bf16(
              false, af[mi], false, bf_[ni], (short)0, acc[mi][ni], false, false);
    }
    buf ^= 1;
  }

  // f32 C/D layout: lanes 0-15: M=v, N=lane ; lanes 16-31: M=v+8, N=lane-16
  int hi8 = (lane >> 4) << 3;
  int nl  = lane & 15;
#pragma unroll
  for (int ni = 0; ni < 4; ++ni) {
    int n = bn + wn * 64 + ni * 16 + nl;
    float bias = ybias[n];
#pragma unroll
    for (int mi = 0; mi < 2; ++mi) {
      int mrow = bm + wm * 32 + mi * 16 + hi8;
#pragma unroll
      for (int v = 0; v < 8; ++v)
        Y[(size_t)(mrow + v) * OUT_DIM + n] = acc[mi][ni][v] + bias;
    }
  }
}

// --------------------------------------------- row LayerNorm + PReLU (in place)
__global__ __launch_bounds__(256) void kan_ln_prelu(float* __restrict__ Y,
    const float* __restrict__ g, const float* __restrict__ be,
    const float* __restrict__ pw) {
  __shared__ float s1[256];
  __shared__ float s2[256];
  float* yr = Y + (size_t)blockIdx.x * OUT_DIM;
  float sum = 0.f, sq = 0.f;
  for (int o = threadIdx.x; o < OUT_DIM; o += 256) {
    float v = yr[o]; sum += v; sq += v * v;
  }
  s1[threadIdx.x] = sum; s2[threadIdx.x] = sq;
  __syncthreads();
  for (int s = 128; s > 0; s >>= 1) {
    if (threadIdx.x < s) { s1[threadIdx.x] += s1[threadIdx.x + s]; s2[threadIdx.x] += s2[threadIdx.x + s]; }
    __syncthreads();
  }
  float mean = s1[0] * (1.f / OUT_DIM);
  float var  = s2[0] * (1.f / OUT_DIM) - mean * mean;
  float rstd = rsqrtf(var + 1e-5f);
  float w = pw[0];
  for (int o = threadIdx.x; o < OUT_DIM; o += 256) {
    float v = (yr[o] - mean) * rstd * g[o] + be[o];
    yr[o] = v > 0.f ? v : w * v;
  }
}

extern "C" void kernel_launch(void* const* d_in, const int* in_sizes, int n_in,
                              void* d_out, int out_size, void* d_ws, size_t ws_size,
                              hipStream_t stream) {
  (void)in_sizes; (void)n_in; (void)out_size; (void)ws_size;
  const float* x            = (const float*)d_in[0];
  const float* spline_w     = (const float*)d_in[1];
  const float* base_w       = (const float*)d_in[2];
  const float* base_bias    = (const float*)d_in[3];
  const float* scale_base   = (const float*)d_in[4];
  const float* scale_spline = (const float*)d_in[5];
  const float* bn_gamma     = (const float*)d_in[6];
  const float* bn_beta      = (const float*)d_in[7];
  const float* ln_gamma     = (const float*)d_in[8];
  const float* ln_beta      = (const float*)d_in[9];
  const float* prelu_w      = (const float*)d_in[10];
  float* out = (float*)d_out;

  char* ws = (char*)d_ws;
  float*    mu    = (float*)(ws);
  float*    rstd  = (float*)(ws + 8192);
  float*    ybias = (float*)(ws + 16384);
  uint16_t* Aact  = (uint16_t*)(ws + 32768);
  uint16_t* Wmat  = (uint16_t*)(ws + 32768 + (size_t)B_DIM * K_DIM * 2);

  kan_colstats<<<dim3(IN_DIM / 32), dim3(32, 8), 0, stream>>>(x, mu, rstd);
  kan_build_act<<<(B_DIM * IN_DIM) / 256, 256, 0, stream>>>(x, mu, rstd, bn_gamma, bn_beta, Aact);
  kan_build_w<<<(OUT_DIM * IN_DIM) / 256, 256, 0, stream>>>(base_w, spline_w, scale_base,
                                                            scale_spline, base_bias, Wmat, ybias);
  kan_gemm<<<dim3(B_DIM / 128, OUT_DIM / 128), 256, 0, stream>>>(Aact, Wmat, ybias, out);
  kan_ln_prelu<<<B_DIM, 256, 0, stream>>>(out, ln_gamma, ln_beta, prelu_w);
}
